// ProsodyExtractor_27968827031676
// MI455X (gfx1250) — compile-verified, matrix operand order checked by reading, not run on hardware
//
#include <hip/hip_runtime.h>
#include <hip/hip_bf16.h>
#include <stdint.h>

typedef __attribute__((ext_vector_type(16))) _Float16 v16h;
typedef __attribute__((ext_vector_type(8)))  float    v8f;
typedef __attribute__((ext_vector_type(4)))  int      v4i;

#define BB 4
#define SS 128
#define NN 512        // B*S segments
#define TT 16         // max_dur
#define CC 80         // mel channels
#define CP 96         // padded to 3x32 for K-chunking
#define HH 256        // d_model
#define G3 768        // 3*H gates
#define KK 9          // conv taps
#define CEPS 1e-5f

#if __has_builtin(__builtin_amdgcn_global_load_async_to_lds_b128)
#define HAVE_ASYNC_LDS 1
typedef __attribute__((address_space(1))) v4i* gv4i_p;
typedef __attribute__((address_space(3))) v4i* lv4i_p;
#else
#define HAVE_ASYNC_LDS 0
#endif

union HF { uint32_t u; _Float16 h[2]; };

__device__ __forceinline__ int kpat(int j, int ko) {
  // 16-bit A-matrix K index pattern for v16h element pair j (ISA 7.12.2)
  return (j < 4) ? (2*j + ko) : (16 + 2*(j-4) + ko);
}

// Build an A/B fragment from a row whose K-pairs are contiguous f16 in memory.
__device__ __forceinline__ v16h frag_row(const _Float16* rowbase, int ko) {
  v16h v;
  const uint32_t* u = (const uint32_t*)rowbase;
  #pragma unroll
  for (int j = 0; j < 8; ++j) {
    int kk = kpat(j, ko);
    HF t; t.u = u[kk >> 1];
    v[2*j]     = t.h[0];
    v[2*j + 1] = t.h[1];
  }
  return v;
}

__device__ __forceinline__ v8f wmma16(v16h a, v16h b, v8f c) {
  return __builtin_amdgcn_wmma_f32_16x16x32_f16(false, a, false, b, (short)0, c,
                                                false, false);
}

__device__ __forceinline__ float sigm(float x) { return 1.0f / (1.0f + __expf(-x)); }

// ---------------------------------------------------------------- prep kernels
__global__ void k_scan(const int* __restrict__ duration, const int* __restrict__ src_len,
                       int* __restrict__ dur, int* __restrict__ starts) {
  int b = threadIdx.x;
  if (b >= BB) return;
  int acc = 0, sl = src_len[b];
  for (int s = 0; s < SS; ++s) {
    int dv = (s < sl) ? duration[b*SS + s] : 0;
    if (dv > TT) dv = TT;
    starts[b*SS + s] = acc;
    dur[b*SS + s]    = dv;
    acc += dv;
  }
}

__global__ void k_w1(const float* __restrict__ w, _Float16* __restrict__ w1t) {
  int idx = blockIdx.x*blockDim.x + threadIdx.x;       // w1t[h][k] (k padded to 16)
  if (idx >= HH*16) return;
  int h = idx >> 4, k = idx & 15;
  w1t[idx] = (k < KK) ? (_Float16)w[k*HH + h] : (_Float16)0.0f;
}

__global__ void k_w2(const float* __restrict__ w, _Float16* __restrict__ w2f) {
  int idx = blockIdx.x*blockDim.x + threadIdx.x;       // w2f[k][h] (k padded to 16)
  if (idx >= 16*HH) return;
  int k = idx >> 8;
  w2f[idx] = (k < KK) ? (_Float16)w[idx - (k<<8) + k*HH] : (_Float16)0.0f;
}

__global__ void k_wih(const float* __restrict__ wf, const float* __restrict__ wb,
                      _Float16* __restrict__ o) {
  int idx = blockIdx.x*blockDim.x + threadIdx.x;       // o[dir][g][c] c padded to 96
  if (idx >= 2*G3*CP) return;
  int dir = idx / (G3*CP), rem = idx % (G3*CP);
  int g = rem / CP, c = rem % CP;
  const float* src = dir ? wb : wf;
  o[idx] = (c < CC) ? (_Float16)src[g*CC + c] : (_Float16)0.0f;
}

__global__ void k_whh(const float* __restrict__ wf, const float* __restrict__ wb,
                      _Float16* __restrict__ o) {
  int idx = blockIdx.x*blockDim.x + threadIdx.x;       // o[dir][g][j]
  if (idx >= 2*G3*HH) return;
  int dir = idx / (G3*HH), rem = idx % (G3*HH);
  o[idx] = (_Float16)((dir ? wb : wf)[rem]);
}

__global__ void k_zero_xg(_Float16* __restrict__ xg) {
  int idx = blockIdx.x*blockDim.x + threadIdx.x;       // zero pad cols [80,96)
  if (idx >= NN*TT*16) return;
  int row = idx >> 4, i = idx & 15;
  xg[(size_t)row*CP + CC + i] = (_Float16)0.0f;
}

__global__ void k_gather(const float* __restrict__ mel, const int* __restrict__ dur,
                         const int* __restrict__ starts, int M,
                         _Float16* __restrict__ segs) {
  int idx = blockIdx.x*blockDim.x + threadIdx.x;       // segs[n][c][t]
  if (idx >= NN*CC*TT) return;
  int t = idx % TT;
  int c = (idx / TT) % CC;
  int n = idx / (CC*TT);
  int b = n / SS;
  float val = 0.0f;
  if (t < dur[n]) {
    int mi = starts[n] + t;
    if (mi > M - 1) mi = M - 1;
    val = mel[((size_t)b*M + mi)*CC + c];
  }
  segs[idx] = (_Float16)val;
}

// --------------------------------------------- fused conv1+BN1+relu+conv2+BN2+relu
// One wave per (n,c) pair. conv1: A1(16x32 im2col over taps) x W1(32x256) -> XC1.
// conv2 re-expressed as E = XC1 x W2^T (16x256x16) then y[t] = sum_k E[t+k-4][k].
// w2f (8KB) is staged block-wide into LDS via async-load-to-LDS, overlapped with
// the conv1 WMMA loop (CDNA5 ASYNCcnt path).
__global__ __launch_bounds__(128) void k_conv(
    const _Float16* __restrict__ segs, const _Float16* __restrict__ w1t,
    const _Float16* __restrict__ w2f, const float* __restrict__ bn1_g,
    const float* __restrict__ bn1_b, const float* __restrict__ bn2_g,
    const float* __restrict__ bn2_b, const int* __restrict__ dur,
    _Float16* __restrict__ xg) {
  __shared__ __align__(16) _Float16 s_xc1[4][TT][HH];
  __shared__ __align__(16) _Float16 s_w2[16][HH];
  __shared__ __align__(16) float    s_e[4][TT][TT];
  const int w    = threadIdx.x >> 5;
  const int lane = threadIdx.x & 31;
  const int m    = lane & 15;
  const int hl   = lane >> 4;
  const int ko   = hl * 8;
  const int p    = blockIdx.x * 4 + w;   // (n,c) pair index
  const int n    = p / CC;
  const int dv   = dur[n];

  // ---- stage w2f[16][256] into LDS (async if available), overlap with conv1
#if HAVE_ASYNC_LDS
  {
    #pragma unroll
    for (int it = 0; it < 4; ++it) {
      int i = it*128 + threadIdx.x;                     // 16-byte chunk index
      __builtin_amdgcn_global_load_async_to_lds_b128(
          (gv4i_p)((v4i*)w2f + i),
          (lv4i_p)((v4i*)&s_w2[0][0] + i),
          0, 0);
    }
  }
#else
  {
    const uint4* src = (const uint4*)w2f;
    uint4* dst = (uint4*)&s_w2[0][0];
    #pragma unroll
    for (int it = 0; it < 4; ++it) {
      int i = it*128 + threadIdx.x;
      dst[i] = src[i];
    }
  }
#endif

  // A1[t=m][k] = x[t + k - 4], zero outside taps/time (segments already zero-padded)
  const _Float16* x = segs + (size_t)p * TT;
  v16h a1;
  #pragma unroll
  for (int j = 0; j < 8; ++j) {
    int kk = kpat(j, ko);
    #pragma unroll
    for (int e2 = 0; e2 < 2; ++e2) {
      int k2 = kk + e2;
      int tt2 = m + k2 - 4;
      _Float16 val = (_Float16)0.0f;
      if (k2 < KK && tt2 >= 0 && tt2 < TT) val = x[tt2];
      a1[2*j + e2] = val;
    }
  }
  const float s1c = rsqrtf(1.0f + CEPS);

  #pragma unroll 1
  for (int ht = 0; ht < 16; ++ht) {
    const int h = ht*16 + m;                            // output channel column
    v16h b1 = frag_row(w1t + h*16, ko);                 // B1[k][h] = w1t[h][k]
    v8f z = {0,0,0,0,0,0,0,0};
    v8f acc = wmma16(a1, b1, z);
    const float sc = bn1_g[h] * s1c;
    const float bt = bn1_b[h];
    #pragma unroll
    for (int v = 0; v < 8; ++v) {
      int row = v + 8*hl;
      float val = fmaxf(acc[v]*sc + bt, 0.0f);
      if (row >= dv) val = 0.0f;                        // re-zero padding
      s_xc1[w][row][h] = (_Float16)val;
    }
  }
#if HAVE_ASYNC_LDS
#if __has_builtin(__builtin_amdgcn_s_wait_asynccnt)
  __builtin_amdgcn_s_wait_asynccnt(0);
#else
  asm volatile("s_wait_asynccnt 0" ::: "memory");
#endif
#endif
  __syncthreads();

  // E = XC1 (16x256) x W2^T (256x16); B2[h][k] = s_w2[k][h]  (LDS-resident)
  v8f e = {0,0,0,0,0,0,0,0};
  const _Float16* arow = &s_xc1[w][m][0];
  const _Float16* brow = &s_w2[m][0];                   // col k = lane&15
  #pragma unroll
  for (int ch = 0; ch < 8; ++ch) {
    v16h a2 = frag_row(arow + ch*32, ko);
    v16h b2 = frag_row(brow + ch*32, ko);
    e = wmma16(a2, b2, e);
  }
  #pragma unroll
  for (int v = 0; v < 8; ++v) s_e[w][v + 8*hl][m] = e[v];
  __syncthreads();

  if (lane < 16) {
    const int t = lane;
    float y = 0.0f;
    #pragma unroll
    for (int k = 0; k < KK; ++k) {
      int s2 = t + k - 4;
      if (s2 >= 0 && s2 < TT) y += s_e[w][s2][k];
    }
    y = fmaxf(y * (bn2_g[0] * s1c) + bn2_b[0], 0.0f);
    if (t >= dv) y = 0.0f;
    xg[((size_t)n*TT + t)*CP + (p - n*CC)] = (_Float16)y;
  }
}

// ---------------------------------------------------- GRU input-gate GEMMs (both dirs)
// G[dir][n][t][768] = x(_rev)[n,t,:] @ W_ih^T + b_ih.  M-tile == one segment (T=16).
__global__ __launch_bounds__(128) void k_gates(
    const _Float16* __restrict__ xg, const _Float16* __restrict__ wih,
    const float* __restrict__ bih_f, const float* __restrict__ bih_b,
    const int* __restrict__ dur, float* __restrict__ G) {
  const int w    = threadIdx.x >> 5;
  const int lane = threadIdx.x & 31;
  const int m = lane & 15, hl = lane >> 4, ko = hl*8;
  int q = blockIdx.x * 4 + w;
  const int dir = q / (NN*48);
  q %= (NN*48);
  const int n = q / 48, nt = q % 48;
  const int t = m;
  int tsrc = t;
  if (dir) {
    int d = dur[n];
    tsrc = d - 1 - t;
    tsrc = tsrc < 0 ? 0 : (tsrc > 15 ? 15 : tsrc);
  }
  const _Float16* xrow = xg + ((size_t)n*TT + tsrc)*CP;
  const int g = nt*16 + m;
  const _Float16* wrow = wih + (size_t)dir*G3*CP + (size_t)g*CP;
  __builtin_prefetch(wrow, 0, 1);
  v8f acc = {0,0,0,0,0,0,0,0};
  #pragma unroll
  for (int ch = 0; ch < 3; ++ch) {
    v16h a = frag_row(xrow + ch*32, ko);
    v16h b = frag_row(wrow + ch*32, ko);
    acc = wmma16(a, b, acc);
  }
  const float bb = (dir ? bih_b : bih_f)[g];
  float* Gp = G + (size_t)dir*NN*TT*G3;
  #pragma unroll
  for (int v = 0; v < 8; ++v) {
    int row = n*16 + v + 8*hl;                          // (n, t=v+8*hl)
    Gp[(size_t)row*G3 + g] = acc[v] + bb;
  }
}

// --------------------------------------------------------- GRU recurrence (one dir)
// One block per 16 segments; h (16x256) lives in LDS as f16; per step:
// hg = h @ W_hh^T via WMMA, gate nonlinearities in registers, masked update.
__global__ __launch_bounds__(256) void k_gru(
    const float* __restrict__ G, const _Float16* __restrict__ whh,
    const float* __restrict__ bhh, const int* __restrict__ dur,
    const int* __restrict__ src_len, float* __restrict__ out, int dir) {
  __shared__ __align__(16) _Float16 s_h[TT][264];       // padded stride vs banks
  __shared__ int s_dur[TT];
  const int tid = threadIdx.x;
  const int w = tid >> 5, lane = tid & 31;
  const int m = lane & 15, hl = lane >> 4, ko = hl*8;
  const int blk = blockIdx.x;

  if (tid < TT) s_dur[tid] = dur[blk*TT + tid];
  for (int i = tid; i < TT*264; i += 256) (&s_h[0][0])[i] = (_Float16)0.0f;
  __syncthreads();

  int durv[8];
  #pragma unroll
  for (int v = 0; v < 8; ++v) durv[v] = s_dur[v + 8*hl];

  const int ht0 = w*2;                                  // each wave owns 2 h-col tiles
  float hout[2][8];
  #pragma unroll
  for (int u = 0; u < 2; ++u)
    #pragma unroll
    for (int v = 0; v < 8; ++v) hout[u][v] = 0.0f;

  #pragma unroll 1
  for (int t = 0; t < TT; ++t) {
    v16h af[8];                                         // full-h A fragments
    #pragma unroll
    for (int ch = 0; ch < 8; ++ch) af[ch] = frag_row(&s_h[m][ch*32], ko);

    #pragma unroll
    for (int u = 0; u < 2; ++u) {
      const int col = (ht0 + u)*16 + m;
      v8f ga[3];
      #pragma unroll
      for (int gate = 0; gate < 3; ++gate) {
        const int gcol = gate*HH + col;
        const _Float16* wrow = whh + (size_t)gcol*HH;
        __builtin_prefetch(wrow, 0, 1);
        v8f acc = {0,0,0,0,0,0,0,0};
        #pragma unroll
        for (int ch = 0; ch < 8; ++ch) {
          v16h b = frag_row(wrow + ch*32, ko);          // B[j][g] = whh[g][j]
          acc = wmma16(af[ch], b, acc);
        }
        ga[gate] = acc;
      }
      const float br = bhh[col], bz = bhh[HH + col], bn_ = bhh[2*HH + col];
      #pragma unroll
      for (int v = 0; v < 8; ++v) {
        const int rown = blk*TT + v + 8*hl;
        const size_t gbase = ((size_t)rown*TT + t)*G3 + col;
        float xr = G[gbase], xz = G[gbase + HH], xn = G[gbase + 2*HH];
        float rg = sigm(xr + ga[0][v] + br);
        float zg = sigm(xz + ga[1][v] + bz);
        float ng = tanhf(xn + rg*(ga[2][v] + bn_));
        float hold = (float)s_h[v + 8*hl][col];
        float hnew = (1.0f - zg)*ng + zg*hold;
        hout[u][v] = (t < durv[v]) ? hnew : hold;
      }
    }
    __syncthreads();
    #pragma unroll
    for (int u = 0; u < 2; ++u) {
      const int col = (ht0 + u)*16 + m;
      #pragma unroll
      for (int v = 0; v < 8; ++v) s_h[v + 8*hl][col] = (_Float16)hout[u][v];
    }
    __syncthreads();
  }

  #pragma unroll
  for (int u = 0; u < 2; ++u) {
    const int col = (ht0 + u)*16 + m;
    #pragma unroll
    for (int v = 0; v < 8; ++v) {
      const int rown = blk*TT + v + 8*hl;
      const int b = rown >> 7, s = rown & 127;
      float val = (s < src_len[b]) ? hout[u][v] : 0.0f;
      out[(size_t)rown*(2*HH) + dir*HH + col] = val;
    }
  }
}

// ------------------------------------------------------------------- launcher
extern "C" void kernel_launch(void* const* d_in, const int* in_sizes, int n_in,
                              void* d_out, int out_size, void* d_ws, size_t ws_size,
                              hipStream_t stream) {
  (void)n_in; (void)out_size; (void)ws_size;
  const float* mel      = (const float*)d_in[0];
  const int*   duration = (const int*)d_in[2];
  const int*   src_len  = (const int*)d_in[3];
  const float* conv1_w  = (const float*)d_in[5];
  const float* bn1_g    = (const float*)d_in[6];
  const float* bn1_b    = (const float*)d_in[7];
  const float* conv2_w  = (const float*)d_in[8];
  const float* bn2_g    = (const float*)d_in[9];
  const float* bn2_b    = (const float*)d_in[10];
  const float* w_ih_f   = (const float*)d_in[11];
  const float* w_hh_f   = (const float*)d_in[12];
  const float* b_ih_f   = (const float*)d_in[13];
  const float* b_hh_f   = (const float*)d_in[14];
  const float* w_ih_b   = (const float*)d_in[15];
  const float* w_hh_b   = (const float*)d_in[16];
  const float* b_ih_b   = (const float*)d_in[17];
  const float* b_hh_b   = (const float*)d_in[18];
  const int M = in_sizes[0] / (BB * CC);

  char* ws = (char*)d_ws;
  int*      dur    = (int*)(ws + 0);              // [512]
  int*      starts = (int*)(ws + 2048);           // [512]
  _Float16* w1t    = (_Float16*)(ws + 4096);      // [256][16]
  _Float16* w2f    = (_Float16*)(ws + 12288);     // [16][256]
  _Float16* wih    = (_Float16*)(ws + 20480);     // [2][768][96]
  _Float16* whh    = (_Float16*)(ws + 315392);    // [2][768][256]
  _Float16* segs   = (_Float16*)(ws + 1101824);   // [512][80][16]
  _Float16* xg     = (_Float16*)(ws + 2412544);   // [512][16][96]
  float*    G      = (float*)(ws + 3985408);      // [2][512][16][768]
  float*    out    = (float*)d_out;               // [512][512]

  k_scan<<<1, 4, 0, stream>>>(duration, src_len, dur, starts);
  k_w1<<<(HH*16 + 255)/256, 256, 0, stream>>>(conv1_w, w1t);
  k_w2<<<(16*HH + 255)/256, 256, 0, stream>>>(conv2_w, w2f);
  k_wih<<<(2*G3*CP + 255)/256, 256, 0, stream>>>(w_ih_f, w_ih_b, wih);
  k_whh<<<(2*G3*HH + 255)/256, 256, 0, stream>>>(w_hh_f, w_hh_b, whh);
  k_zero_xg<<<(NN*TT*16 + 255)/256, 256, 0, stream>>>(xg);
  k_gather<<<(NN*CC*TT + 255)/256, 256, 0, stream>>>(mel, dur, starts, M, segs);
  k_conv<<<(NN*CC)/4, 128, 0, stream>>>(segs, w1t, w2f, bn1_g, bn1_b,
                                        bn2_g, bn2_b, dur, xg);
  k_gates<<<(2*NN*48)/4, 128, 0, stream>>>(xg, wih, b_ih_f, b_ih_b, dur, G);
  k_gru<<<NN/TT, 256, 0, stream>>>(G, whh, b_hh_f, dur, src_len, out, 0);
  k_gru<<<NN/TT, 256, 0, stream>>>(G + (size_t)NN*TT*G3, whh + (size_t)G3*HH,
                                   b_hh_b, dur, src_len, out, 1);
}
